// PcConvBp_DS_42107859370366
// MI455X (gfx1250) — compile-verified, matrix operand order checked by reading, not run on hardware
//
#include <hip/hip_runtime.h>

typedef __attribute__((ext_vector_type(16))) __bf16 v16bf;
typedef __attribute__((ext_vector_type(8)))  __bf16 v8bf;
typedef __attribute__((ext_vector_type(8)))  float  v8f;
typedef unsigned int u32;
typedef __attribute__((ext_vector_type(4))) u32 u32x4;
typedef __attribute__((ext_vector_type(8))) int i32x8;
typedef __attribute__((ext_vector_type(4))) int i32x4;

#if __has_builtin(__builtin_amdgcn_tensor_load_to_lds)
#define HAVE_TDM 1
#else
#define HAVE_TDM 0
#endif

#define LDS_STRIDE 40   // bf16 elems/row: 80B -> conflict-free across 16 rows, 16B aligned
#define NT 128          // pixel tile per workgroup
#define NTHREADS 256    // 8 waves

// ---------------------------------------------------------------------------
// Implicit-GEMM 3x3 / 1x1 conv with bf16 WMMA, f32 accumulate.
// M = 64 output channels. Each of 8 waves computes a 16(M) x 64(N) D-tile
// (4 accumulators of 16x16). K processed in chunks of 32 through LDS.
// A (weights) arrives pre-converted bf16, chunk-major [chunk][m][32]; each
// chunk is DMA'd into LDS by the Tensor Data Mover with hardware row padding
// (32 elems data + 8 elems pad = stride 40). B (im2col patches) is gathered
// and converted f32->bf16 by the waves while the TDM runs.
// ---------------------------------------------------------------------------
template <int KSZ>
__global__ __launch_bounds__(NTHREADS) void pc_conv_wmma(
    const float*  __restrict__ in,    // input activations (f32)
    const __bf16* __restrict__ wgtc,  // weights, bf16 chunk-major [ktot/32][64][32]
    float*        __restrict__ outp,  // output buffer (f32)
    const float*  __restrict__ aux,   // mode1: x ; mode3: y_pad ; else unused
    float*        __restrict__ partials, // mode1: per-block sum of squares
    const float*  __restrict__ sumsq,    // mode2: ||r||^2 scalar
    int Hi, int Wi, int pad,          // logical input spatial dims, border pad
    int ktot,                         // 576 (3x3) or 64 (1x1)
    int Ho, int Wo,                   // logical output spatial dims
    int outW, int outChan, int outBatch, int outOff,
    int auxW, int auxChan, int auxBatch, int auxOff,
    int mode, int numN)
{
    __shared__ __bf16 ldsA[64 * LDS_STRIDE];
    __shared__ __bf16 ldsB[NT * LDS_STRIDE];
    __shared__ float  red[NTHREADS];

    constexpr int KS2 = KSZ * KSZ;

    const int tid  = threadIdx.x;
    const int lane = tid & 31;
    const int wave = tid >> 5;
    const int m0   = (wave & 3) * 16;      // M block of this wave
    const int nwb  = (wave >> 2) * 64;     // N sub-tile base of this wave

    const int inChan  = Hi * Wi;
    const int inBatch = 64 * inChan;
    const int ntile   = blockIdx.x * NT;

    // B-stage coordinates (constant across the K loop): (nl = tid&127, kl = tid>>7 + 2t)
    const int nlB   = tid & 127;
    const int klB0  = tid >> 7;
    const int nSt   = ntile + nlB;
    const bool nStValid = (nSt < numN);
    int ohS = 0, owS = 0, inBase = 0;
    if (nStValid) {
        int b = nSt / (Ho * Wo);
        int r = nSt - b * (Ho * Wo);
        ohS = r / Wo;
        owS = r - ohS * Wo;
        inBase = b * inBatch;
    }

#if HAVE_TDM
    // TDM descriptor constants: 1-D tensor of 2048 bf16 (one 64x32 A chunk),
    // LDS padding of 16B after every 64B -> LDS rows of stride 40 bf16.
    const u32 ldsAoff = (u32)(unsigned long long)(uintptr_t)&ldsA[0]; // LDS byte offset
    const unsigned long long gbase = (unsigned long long)(uintptr_t)wgtc;
    const int g1w0 = (int)((1u << 16) |      // data_size = 2 bytes
                           (1u << 20) |      // pad_enable
                           (3u << 22) |      // pad_interval: 64B of data
                           (3u << 25));      // pad_amount: 4 DWORDs (16B)
#else
    const int klA = tid & 31;                // manual A-stage coords
    const int mA  = tid >> 5;
#endif

    v8f acc[4] = {};

    for (int kb = 0; kb < ktot; kb += 32) {
        __syncthreads();
#if HAVE_TDM
        if (wave == 0) {
            unsigned long long ga = gbase + (unsigned long long)kb * 128ull; // kb*64 elems * 2B
            u32x4 g0 = { 1u,                                   // count=1, user descriptor
                         ldsAoff,                              // lds_addr
                         (u32)ga,                              // global_addr[31:0]
                         (u32)((ga >> 32) & 0x01FFFFFFu) | (2u << 30) }; // ga[56:32], type=2
            i32x8 g1 = { g1w0,
                         (int)0x08000000u,    // [31:16] tensor_dim0.lo = 2048
                         (int)(1 << 16),      // tensor_dim0.hi=0, tensor_dim1.lo=1
                         (int)(2048u << 16),  // tensor_dim1.hi=0, tile_dim0=2048
                         1,                   // tile_dim1=1, tile_dim2=0
                         2048,                // tensor_dim0_stride[31:0]
                         (int)(2048u << 16),  // stride0.hi=0, tensor_dim1_stride.lo=2048
                         0 };
            i32x4 z4 = { 0, 0, 0, 0 };
            i32x8 z8 = { 0, 0, 0, 0, 0, 0, 0, 0 };
            __builtin_amdgcn_tensor_load_to_lds(g0, g1, z4, z4, z8, 0);
        }
#else
        #pragma unroll
        for (int t = 0; t < 8; ++t) {
            int m = mA + 8 * t;
            ldsA[m * LDS_STRIDE + klA] = wgtc[kb * 64 + m * 32 + klA];
        }
#endif
        // ---- stage B (im2col patches) : NT x 32, f32 -> bf16 (overlaps TDM) ----
        #pragma unroll
        for (int t = 0; t < 16; ++t) {
            int kl = klB0 + 2 * t;
            int k  = kb + kl;
            int ci = k / KS2;
            int rm = k - ci * KS2;
            int u  = rm / KSZ;
            int vv = rm - u * KSZ;
            float v = 0.f;
            if (nStValid) {
                int ih = ohS + u - pad;
                int iw = owS + vv - pad;
                if (ih >= 0 && ih < Hi && iw >= 0 && iw < Wi)
                    v = in[inBase + ci * inChan + ih * Wi + iw];
            }
            ldsB[nlB * LDS_STRIDE + kl] = (__bf16)v;
        }
#if HAVE_TDM
        if (wave == 0) __builtin_amdgcn_s_wait_tensorcnt(0);
#endif
        __syncthreads();

        // ---- fragments + WMMA ----
        const int rsel  = (lane < 16) ? 0 : 8;   // K-half selection per ISA layout
        const int row16 = lane & 15;
        const __bf16* pa = &ldsA[(m0 + row16) * LDS_STRIDE];
        v8bf alo = *(const v8bf*)(pa + rsel);
        v8bf ahi = *(const v8bf*)(pa + rsel + 16);
        v16bf afrag = __builtin_shufflevector(alo, ahi,
            0,1,2,3,4,5,6,7,8,9,10,11,12,13,14,15);

        #pragma unroll
        for (int nb = 0; nb < 4; ++nb) {
            const __bf16* pb = &ldsB[(nwb + nb * 16 + row16) * LDS_STRIDE];
            v8bf blo = *(const v8bf*)(pb + rsel);
            v8bf bhi = *(const v8bf*)(pb + rsel + 16);
            v16bf bfrag = __builtin_shufflevector(blo, bhi,
                0,1,2,3,4,5,6,7,8,9,10,11,12,13,14,15);
            acc[nb] = __builtin_amdgcn_wmma_f32_16x16x32_bf16(
                false, afrag, false, bfrag, (short)0, acc[nb], false, false);
        }
    }

    // ---- epilogue ----
    float scale = 1.f;
    if (mode == 2) scale = 0.01f * rsqrtf(*sumsq);   // LR / ||r||
    float localsq = 0.f;
    const int mtop = (lane >= 16) ? 8 : 0;           // D layout: VGPR j -> M = j (+8)
    #pragma unroll
    for (int nb = 0; nb < 4; ++nb) {
        int n = ntile + nwb + nb * 16 + (lane & 15);
        if (n < numN) {
            int b  = n / (Ho * Wo);
            int r2 = n - b * (Ho * Wo);
            int oh = r2 / Wo;
            int ow = r2 - oh * Wo;
            int obase = b * outBatch + oh * outW + ow + outOff;
            int abase = b * auxBatch + oh * auxW + ow + auxOff;
            #pragma unroll
            for (int j = 0; j < 8; ++j) {
                int co = m0 + mtop + j;
                float v = acc[nb][j];
                int oi = obase + co * outChan;
                if (mode == 0) {
                    outp[oi] = v > 0.f ? v : 0.f;              // ReLU(FF conv)
                } else if (mode == 1) {
                    float r = aux[abase + co * auxChan] - v;    // r = x - conv(y_pad)
                    outp[oi] = r;
                    localsq += r * r;
                } else if (mode == 2) {
                    outp[oi] += scale * v;                      // y_pad += lr*g
                } else {
                    outp[oi] = v + aux[abase + co * auxChan];   // bypass + y_opt
                }
            }
        }
    }

    if (mode == 1) {    // deterministic per-block reduction
        red[tid] = localsq;
        __syncthreads();
        #pragma unroll
        for (int s = NTHREADS / 2; s > 0; s >>= 1) {
            if (tid < s) red[tid] += red[tid + s];
            __syncthreads();
        }
        if (tid == 0) partials[blockIdx.x] = red[0];
    }
}

// ---------------------------------------------------------------------------
// Convert all weights to bf16, chunk-major [chunk][m][32] (4KB per chunk, TDM
// friendly). Also builds the flipped/channel-transposed feedback weight:
//   wfbt[m][i*9+u*3+v] = w_fb[i][m][2-u][2-v]
// ---------------------------------------------------------------------------
__global__ void build_weights(const float* __restrict__ wff,
                              const float* __restrict__ wfb,
                              const float* __restrict__ wby,
                              __bf16* __restrict__ wffc,
                              __bf16* __restrict__ wfbc,
                              __bf16* __restrict__ wfbtc,
                              __bf16* __restrict__ wbyc)
{
    int idx = blockIdx.x * 256 + threadIdx.x;
    if (idx < 64 * 576) {
        int chunk = idx >> 11;        // / 2048
        int r     = idx & 2047;
        int m  = r >> 5, kl = r & 31;
        int k  = chunk * 32 + kl;
        wffc[idx] = (__bf16)wff[m * 576 + k];
        wfbc[idx] = (__bf16)wfb[m * 576 + k];
        int i = k / 9, rm = k - i * 9;
        int u = rm / 3, v = rm - u * 3;
        wfbtc[idx] = (__bf16)wfb[i * 576 + m * 9 + (2 - u) * 3 + (2 - v)];
    }
    if (idx < 64 * 64) {
        int chunk = idx >> 11;
        int r     = idx & 2047;
        int m  = r >> 5, kl = r & 31;
        wbyc[idx] = (__bf16)wby[m * 64 + chunk * 32 + kl];
    }
}

// Deterministic fixed-order final reduction of block partials.
__global__ void reduce_partials(const float* __restrict__ partials, int n,
                                float* __restrict__ sumsq) {
    __shared__ float red[256];
    float s = 0.f;
    for (int i = threadIdx.x; i < n; i += 256) s += partials[i];
    red[threadIdx.x] = s;
    __syncthreads();
    for (int st = 128; st > 0; st >>= 1) {
        if (threadIdx.x < st) red[threadIdx.x] += red[threadIdx.x + st];
        __syncthreads();
    }
    if (threadIdx.x == 0) *sumsq = red[0];
}

// ---------------------------------------------------------------------------
extern "C" void kernel_launch(void* const* d_in, const int* in_sizes, int n_in,
                              void* d_out, int out_size, void* d_ws, size_t ws_size,
                              hipStream_t stream) {
    const float* x    = (const float*)d_in[0];   // [16,64,112,112]
    const float* w_ff = (const float*)d_in[1];   // [64,64,3,3]
    const float* w_fb = (const float*)d_in[2];   // [64,64,3,3]
    const float* w_by = (const float*)d_in[3];   // [64,64,1,1]
    float* out = (float*)d_out;

    // workspace layout (bytes, all 256-aligned)
    const size_t YPAD_B = (size_t)16 * 64 * 114 * 114 * 4;   // 53,231,616
    const size_t RPAD_B = (size_t)16 * 64 * 116 * 116 * 4;   // 55,115,776
    const size_t WC_B   = (size_t)64 * 576 * 2;              // 73,728 (bf16)
    const size_t WBY_B  = (size_t)64 * 64 * 2;               // 8,192  (bf16)
    const size_t PART_B = 4096 * 4;
    char* ws = (char*)d_ws;
    size_t off = 0;
    float*  ypad  = (float*) (ws + off); off += YPAD_B;
    float*  rpad  = (float*) (ws + off); off += RPAD_B;
    __bf16* wffc  = (__bf16*)(ws + off); off += WC_B;
    __bf16* wfbc  = (__bf16*)(ws + off); off += WC_B;
    __bf16* wfbtc = (__bf16*)(ws + off); off += WC_B;
    __bf16* wbyc  = (__bf16*)(ws + off); off += WBY_B;
    float* partials = (float*)(ws + off); off += PART_B;
    float* sumsq    = (float*)(ws + off); off += 256;
    if (ws_size < off) return;

    // zero padded buffers (borders must be 0; interiors rewritten below)
    (void)hipMemsetAsync(ypad, 0, YPAD_B, stream);
    (void)hipMemsetAsync(rpad, 0, RPAD_B, stream);
    build_weights<<<(64 * 576 + 255) / 256, 256, 0, stream>>>(
        w_ff, w_fb, w_by, wffc, wfbc, wfbtc, wbyc);

    const int N1 = 16 * 112 * 112;           // 200704 output pixels (112x112 convs)
    const int g1 = (N1 + NT - 1) / NT;       // 1568
    const int N3 = 16 * 114 * 114;           // 207936 (full padded-code update)
    const int g3 = (N3 + NT - 1) / NT;       // 1625

    // K1: y_pad[interior] = relu(conv_same(x, w_ff))
    pc_conv_wmma<3><<<g1, NTHREADS, 0, stream>>>(
        x, wffc, ypad, nullptr, nullptr, nullptr,
        /*Hi,Wi,pad*/ 112, 112, 1, /*ktot*/ 576, /*Ho,Wo*/ 112, 112,
        /*out*/ 114, 114 * 114, 64 * 114 * 114, 114 + 1,
        /*aux*/ 0, 0, 0, 0, /*mode*/ 0, N1);

    for (int it = 0; it < 5; ++it) {
        // K2: r_pad[interior] = x - conv_valid(y_pad, w_fb); partial sums of r^2
        pc_conv_wmma<3><<<g1, NTHREADS, 0, stream>>>(
            ypad, wfbc, rpad, x, partials, nullptr,
            114, 114, 0, 576, 112, 112,
            /*out*/ 116, 116 * 116, 64 * 116 * 116, 2 * 116 + 2,
            /*aux(x)*/ 112, 112 * 112, 64 * 112 * 112, 0, /*mode*/ 1, N1);
        reduce_partials<<<1, 256, 0, stream>>>(partials, g1, sumsq);
        // K3: y_pad += (LR/||r||) * conv_valid(r_pad, flipT(w_fb))   [full 114x114]
        pc_conv_wmma<3><<<g3, NTHREADS, 0, stream>>>(
            rpad, wfbtc, ypad, nullptr, nullptr, sumsq,
            116, 116, 0, 576, 114, 114,
            /*out*/ 114, 114 * 114, 64 * 114 * 114, 0,
            0, 0, 0, 0, /*mode*/ 2, N3);
    }

    // K4: out = y_pad[interior] + conv1x1(x, w_bypass)
    pc_conv_wmma<1><<<g1, NTHREADS, 0, stream>>>(
        x, wbyc, out, ypad, nullptr, nullptr,
        112, 112, 0, /*ktot*/ 64, 112, 112,
        /*out*/ 112, 112 * 112, 64 * 112 * 112, 0,
        /*aux(y_pad)*/ 114, 114 * 114, 64 * 114 * 114, 114 + 1, /*mode*/ 3, N1);
}
